// Videollama3VisionAttention_28750511079650
// MI455X (gfx1250) — matrix operand order repaired and checked by reading, not compiled, not run on hardware
//
#include <hip/hip_runtime.h>
#include <hip/hip_bf16.h>

// ---------------------------------------------------------------------------
// Videollama3 vision attention for gfx1250 (MI455X).
// f16 WMMA (V_WMMA_F32_16X16X32_F16) everywhere, f32 accumulate.
// Pipeline: cvt(f32->f16) -> QKV GEMMs -> RoPE repack -> flash attention
//           -> O-projection GEMM.
// ---------------------------------------------------------------------------

typedef __attribute__((ext_vector_type(16))) _Float16 v16h;
typedef __attribute__((ext_vector_type(8)))  _Float16 v8h;
typedef __attribute__((ext_vector_type(8)))  float    v8f;

static constexpr int S   = 2048;   // packed tokens
static constexpr int E   = 1152;   // hidden size
static constexpr int H   = 16;     // heads
static constexpr int D   = 72;     // head dim
static constexpr int DP  = 96;     // padded head dim (3 * 32 WMMA K-steps)

// ===========================================================================
// Elementwise f32 -> f16 conversion (weights + activations, once per call)
// ===========================================================================
__global__ __launch_bounds__(256)
void cvt_f16_kernel(const float* __restrict__ src, _Float16* __restrict__ dst,
                    int n) {
  const int i = blockIdx.x * 256 + threadIdx.x;
  if (i < n) dst[i] = (_Float16)src[i];
}

// ===========================================================================
// Tiled GEMM: C[M,N] = A[M,K] * B[N,K]^T + bias[N]; A,B f16 row-major.
// Block = 128 threads = 4 waves; block tile 128x64; wave tile 32x64
// (2 A-fragments x 4 B-fragments = 8 WMMAs per 32-wide K step).
// Register-prefetch of the next K tile overlaps global loads with WMMAs.
// ===========================================================================
__global__ __launch_bounds__(128)
void gemm_f16_bias_kernel(const _Float16* __restrict__ A,
                          const _Float16* __restrict__ B,
                          const float* __restrict__ bias,
                          float* __restrict__ C,
                          int M, int N, int K) {
  __shared__ alignas(16) _Float16 As[128][40];  // [row][k], padded stride
  __shared__ alignas(16) _Float16 Bs[64][40];   // [n][k]

  const int tid  = threadIdx.x;
  const int lane = tid & 31;
  const int wave = tid >> 5;
  const int m0   = blockIdx.y * 128;
  const int n0   = blockIdx.x * 64;

  // staging map: 8-half (16B) chunks. A: 512 chunks -> 4/thread; B: 256 -> 2.
  int aRow[4], aCol[4], bRow[2], bCol[2];
#pragma unroll
  for (int j = 0; j < 4; ++j) {
    const int c = tid + j * 128;       // chunk id 0..511
    aRow[j] = c >> 2;                  // 4 chunks per 32-col row
    aCol[j] = (c & 3) * 8;
  }
#pragma unroll
  for (int j = 0; j < 2; ++j) {
    const int c = tid + j * 128;       // chunk id 0..255
    bRow[j] = c >> 2;
    bCol[j] = (c & 3) * 8;
  }

  v8f acc[2][4] = {};

  // ---- prologue: stage K tile 0 -------------------------------------------
  {
    v8h ra[4], rb[2];
#pragma unroll
    for (int j = 0; j < 4; ++j)
      ra[j] = *(const v8h*)(A + (size_t)(m0 + aRow[j]) * K + aCol[j]);
#pragma unroll
    for (int j = 0; j < 2; ++j)
      rb[j] = *(const v8h*)(B + (size_t)(n0 + bRow[j]) * K + bCol[j]);
#pragma unroll
    for (int j = 0; j < 4; ++j) *(v8h*)&As[aRow[j]][aCol[j]] = ra[j];
#pragma unroll
    for (int j = 0; j < 2; ++j) *(v8h*)&Bs[bRow[j]][bCol[j]] = rb[j];
  }
  __syncthreads();

  const int kh = (lane >> 4) * 8;
  const int kb = (lane >> 4) * 16;

  for (int k0 = 0; k0 < K; k0 += 32) {
    const bool hasNext = (k0 + 32) < K;

    // ---- prefetch next K tile into registers (overlaps with WMMAs) -------
    v8h ra[4], rb[2];
    if (hasNext) {
      const int kn = k0 + 32;
#pragma unroll
      for (int j = 0; j < 4; ++j)
        ra[j] = *(const v8h*)(A + (size_t)(m0 + aRow[j]) * K + kn + aCol[j]);
#pragma unroll
      for (int j = 0; j < 2; ++j)
        rb[j] = *(const v8h*)(B + (size_t)(n0 + bRow[j]) * K + kn + bCol[j]);
    }

    // ---- A fragments: rows wave*32 + mi*16 + (lane&15) --------------------
    // lane 0-15 : K = 0..15; lane 16-31: K = 16..31
    v16h a[2];
#pragma unroll
    for (int mi = 0; mi < 2; ++mi) {
      const int mrow = wave * 32 + mi * 16 + (lane & 15);
#pragma unroll
      for (int i = 0; i < 16; ++i) a[mi][i] = As[mrow][kb + i];
    }

    // ---- 4 N-subtiles x 2 M-subtiles --------------------------------------
#pragma unroll
    for (int nt = 0; nt < 4; ++nt) {
      v16h b;
      const int nrow = nt * 16 + (lane & 15);
#pragma unroll
      for (int i = 0; i < 16; ++i) b[i] = Bs[nrow][kb + i];
#pragma unroll
      for (int mi = 0; mi < 2; ++mi)
        acc[mi][nt] = __builtin_amdgcn_wmma_f32_16x16x32_f16(
            false, a[mi], false, b, (short)0, acc[mi][nt], false, false);
    }

    __syncthreads();                 // all waves done reading this tile
    if (hasNext) {
#pragma unroll
      for (int j = 0; j < 4; ++j) *(v8h*)&As[aRow[j]][aCol[j]] = ra[j];
#pragma unroll
      for (int j = 0; j < 2; ++j) *(v8h*)&Bs[bRow[j]][bCol[j]] = rb[j];
    }
    __syncthreads();                 // new tile visible
  }

  // ---- epilogue: C layout row = v + 8*(lane>=16), col = lane&15 -----------
#pragma unroll
  for (int nt = 0; nt < 4; ++nt) {
    const int col = n0 + nt * 16 + (lane & 15);
    const float bv = bias[col];
#pragma unroll
    for (int mi = 0; mi < 2; ++mi) {
#pragma unroll
      for (int v = 0; v < 8; ++v) {
        const int row = m0 + wave * 32 + mi * 16 + v + 8 * (lane >> 4);
        C[(size_t)row * N + col] = acc[mi][nt][v] + bv;
      }
    }
  }
}

// ===========================================================================
// RoPE (f32) + repack to WMMA-friendly f16 layouts:
//   Qp, Kp : [H][S][DP]   (row per token, zero-padded 72->96)
//   Vt     : [H][DP][S]   (transposed so PV B-fragments are contiguous)
// Q is pre-scaled by D^-0.5 so the attention loop skips the score scaling.
// ===========================================================================
__global__ __launch_bounds__(DP)
void rope_repack_kernel(const float* __restrict__ qf,
                        const float* __restrict__ kf,
                        const float* __restrict__ vf,
                        const float* __restrict__ cosb,
                        const float* __restrict__ sinb,
                        _Float16* __restrict__ Qp,
                        _Float16* __restrict__ Kp,
                        _Float16* __restrict__ Vt) {
  const int s = blockIdx.x;
  const int h = blockIdx.y;
  const int d = threadIdx.x;            // 0..95

  _Float16 qo = (_Float16)0.f, ko = (_Float16)0.f, vo = (_Float16)0.f;
  if (d < D) {
    const float scaling = rsqrtf((float)D);
    const size_t base = (size_t)s * E + h * D;
    const float c  = cosb[s * D + d];
    const float sn = sinb[s * D + d];
    const int   hl = D / 2;
    const int   pd = (d < hl) ? d + hl : d - hl;
    const float qv = qf[base + d],  kv = kf[base + d];
    const float qp = qf[base + pd], kp = kf[base + pd];
    const float qr = (d < hl) ? -qp : qp;
    const float kr = (d < hl) ? -kp : kp;
    qo = (_Float16)((qv * c + qr * sn) * scaling);
    ko = (_Float16)(kv * c + kr * sn);
    vo = (_Float16)vf[base + d];
  }
  Qp[((size_t)h * S + s) * DP + d] = qo;
  Kp[((size_t)h * S + s) * DP + d] = ko;
  Vt[((size_t)h * DP + d) * S + s] = vo;
}

// ===========================================================================
// Block-diagonal flash attention. Block = 128 threads = 4 waves.
// Wave owns 16 query rows; grid = (S/64, H). Sk tiles of 32 tokens:
//   6 WMMAs QK^T (3 d-chunks x 2 token halves) -> online softmax ->
//   P via LDS relayout -> 6 WMMAs P*V over the 96-wide padded value dim.
// Output context: f16 [S][H*D].
// ===========================================================================
__global__ __launch_bounds__(128)
void attn_kernel(const _Float16* __restrict__ Qp,
                 const _Float16* __restrict__ Kp,
                 const _Float16* __restrict__ Vt,
                 const int* __restrict__ cu, int nseg,
                 _Float16* __restrict__ ctx) {
  __shared__ _Float16 Pl[4][16][40];     // per-wave P tile [16 rows][32 cols]

  const int lane = threadIdx.x & 31;
  const int wave = threadIdx.x >> 5;
  const int h    = blockIdx.y;
  const int q0   = blockIdx.x * 64 + wave * 16;

  // segment bounds (cu_seqlens boundaries are >= 64-aligned in this model)
  int segLo = 0, segHi = S;
  for (int i = 0; i < nseg; ++i) {
    const int lo = cu[i], hi = cu[i + 1];
    if (lo <= q0 && q0 < hi) { segLo = lo; segHi = hi; }
  }

  const int kh = (lane >> 4) * 8;
  const int kb = (lane >> 4) * 16;

  // ---- Q fragments (3 x 16x32 over padded head dim 96), kept in VGPRs ----
  v16h qa[3];
  {
    const _Float16* qb = Qp + ((size_t)h * S + q0 + (lane & 15)) * DP;
#pragma unroll
    for (int f = 0; f < 3; ++f) {
#pragma unroll
      for (int i = 0; i < 16; ++i) qa[f][i] = qb[f * 32 + kb + i];
    }
  }

  float mrow[8], lrow[8];
#pragma unroll
  for (int v = 0; v < 8; ++v) { mrow[v] = -1e30f; lrow[v] = 0.f; }
  v8f cacc[6] = {};

  for (int kt = segLo; kt < segHi; kt += 32) {
    // ---- scores S = Q K^T for 32 key tokens (Q pre-scaled) ----------------
    v8f s0 = {}, s1 = {};
#pragma unroll
    for (int f = 0; f < 3; ++f) {
      v16h b0, b1;
      const _Float16* k0p =
          Kp + ((size_t)h * S + kt + (lane & 15)) * DP + f * 32 + kb;
      const _Float16* k1p =
          Kp + ((size_t)h * S + kt + 16 + (lane & 15)) * DP + f * 32 + kb;
#pragma unroll
      for (int i = 0; i < 16; ++i) { b0[i] = k0p[i]; b1[i] = k1p[i]; }
      s0 = __builtin_amdgcn_wmma_f32_16x16x32_f16(
          false, qa[f], false, b0, (short)0, s0, false, false);
      s1 = __builtin_amdgcn_wmma_f32_16x16x32_f16(
          false, qa[f], false, b1, (short)0, s1, false, false);
    }

    // ---- online softmax (rows live on 16-lane halves: shfl_xor 1/2/4/8) ---
#pragma unroll
    for (int v = 0; v < 8; ++v) {
      const float x0 = s0[v];
      const float x1 = s1[v];
      float rmax = fmaxf(x0, x1);
#pragma unroll
      for (int m = 1; m < 16; m <<= 1) rmax = fmaxf(rmax, __shfl_xor(rmax, m, 32));
      const float mnew  = fmaxf(mrow[v], rmax);
      const float alpha = __expf(mrow[v] - mnew);
      const float p0 = __expf(x0 - mnew);
      const float p1 = __expf(x1 - mnew);
      float rsum = p0 + p1;
#pragma unroll
      for (int m = 1; m < 16; m <<= 1) rsum += __shfl_xor(rsum, m, 32);
      lrow[v] = lrow[v] * alpha + rsum;
      mrow[v] = mnew;
#pragma unroll
      for (int nt = 0; nt < 6; ++nt) cacc[nt][v] *= alpha;

      const int prow = v + 8 * (lane >> 4);
      Pl[wave][prow][lane & 15]        = (_Float16)p0;
      Pl[wave][prow][16 + (lane & 15)] = (_Float16)p1;
    }
    __syncthreads();   // order LDS stores (C-layout) before A-layout reads

    // ---- reload P as an A fragment ----------------------------------------
    v16h pa;
    const int pm = lane & 15;
#pragma unroll
    for (int i = 0; i < 16; ++i) pa[i] = Pl[wave][pm][kb + i];

    // ---- context += P * V over 6 value-dim subtiles -----------------------
#pragma unroll
    for (int nt = 0; nt < 6; ++nt) {
      v16h vb;
      const _Float16* vp =
          Vt + ((size_t)h * DP + nt * 16 + (lane & 15)) * S + kt + kb;
#pragma unroll
      for (int i = 0; i < 16; ++i) vb[i] = vp[i];
      cacc[nt] = __builtin_amdgcn_wmma_f32_16x16x32_f16(
          false, pa, false, vb, (short)0, cacc[nt], false, false);
    }
    __syncthreads();
  }

  // ---- normalize + write f16 context [S][H*D] (drop pad cols >= 72) -------
#pragma unroll
  for (int v = 0; v < 8; ++v) {
    const float inv = 1.f / lrow[v];
    const int row = q0 + v + 8 * (lane >> 4);
#pragma unroll
    for (int nt = 0; nt < 6; ++nt) {
      const int col = nt * 16 + (lane & 15);
      if (col < D)
        ctx[((size_t)row * H + h) * D + col] = (_Float16)(cacc[nt][v] * inv);
    }
  }
}

// ===========================================================================
// Host-side launch
// ===========================================================================
extern "C" void kernel_launch(void* const* d_in, const int* in_sizes, int n_in,
                              void* d_out, int out_size, void* d_ws, size_t ws_size,
                              hipStream_t stream) {
  const float* hs  = (const float*)d_in[0];
  const float* q_w = (const float*)d_in[1];
  const float* q_b = (const float*)d_in[2];
  const float* k_w = (const float*)d_in[3];
  const float* k_b = (const float*)d_in[4];
  const float* v_w = (const float*)d_in[5];
  const float* v_b = (const float*)d_in[6];
  const float* o_w = (const float*)d_in[7];
  const float* o_b = (const float*)d_in[8];
  const float* cosb = (const float*)d_in[9];
  const float* sinb = (const float*)d_in[10];
  const int*   cu  = (const int*)d_in[11];
  const int nseg = in_sizes[11] - 1;

  // ---- workspace layout ----------------------------------------------------
  char* w = (char*)d_ws;
  size_t off = 0;
  _Float16* hs_h = (_Float16*)(w + off); off += (size_t)S * E * 2;
  _Float16* qw_h = (_Float16*)(w + off); off += (size_t)E * E * 2;
  _Float16* kw_h = (_Float16*)(w + off); off += (size_t)E * E * 2;
  _Float16* vw_h = (_Float16*)(w + off); off += (size_t)E * E * 2;
  _Float16* ow_h = (_Float16*)(w + off); off += (size_t)E * E * 2;
  float* qf = (float*)(w + off);
  float* kf = qf + (size_t)S * E;
  float* vf = kf + (size_t)S * E;
  _Float16* ctxh = (_Float16*)qf;        // aliases qf (dead after repack)
  off += 3u * (size_t)S * E * 4;
  _Float16* Qp = (_Float16*)(w + off); off += (size_t)H * S * DP * 2;
  _Float16* Kp = (_Float16*)(w + off); off += (size_t)H * S * DP * 2;
  _Float16* Vt = (_Float16*)(w + off);

  // 0) one-shot f32 -> f16 conversion of activations + weights
  const int nHS = S * E, nW = E * E;
  cvt_f16_kernel<<<(nHS + 255) / 256, 256, 0, stream>>>(hs,  hs_h, nHS);
  cvt_f16_kernel<<<(nW  + 255) / 256, 256, 0, stream>>>(q_w, qw_h, nW);
  cvt_f16_kernel<<<(nW  + 255) / 256, 256, 0, stream>>>(k_w, kw_h, nW);
  cvt_f16_kernel<<<(nW  + 255) / 256, 256, 0, stream>>>(v_w, vw_h, nW);
  cvt_f16_kernel<<<(nW  + 255) / 256, 256, 0, stream>>>(o_w, ow_h, nW);

  const dim3 gGemm(E / 64, S / 128);     // (18, 16)
  const dim3 bGemm(128);

  // 1) QKV projections
  gemm_f16_bias_kernel<<<gGemm, bGemm, 0, stream>>>(hs_h, qw_h, q_b, qf, S, E, E);
  gemm_f16_bias_kernel<<<gGemm, bGemm, 0, stream>>>(hs_h, kw_h, k_b, kf, S, E, E);
  gemm_f16_bias_kernel<<<gGemm, bGemm, 0, stream>>>(hs_h, vw_h, v_b, vf, S, E, E);

  // 2) RoPE + repack to padded f16 layouts (Q pre-scaled by D^-0.5)
  rope_repack_kernel<<<dim3(S, H), dim3(DP), 0, stream>>>(qf, kf, vf, cosb, sinb,
                                                          Qp, Kp, Vt);

  // 3) block-diagonal flash attention
  attn_kernel<<<dim3(S / 64, H), dim3(128), 0, stream>>>(Qp, Kp, Vt, cu, nseg, ctxh);

  // 4) output projection -> f32 d_out
  gemm_f16_bias_kernel<<<gGemm, bGemm, 0, stream>>>(ctxh, ow_h, o_b,
                                                    (float*)d_out, S, E, E);
}
